// GATLayer_38800734552445
// MI455X (gfx1250) — matrix operand order repaired and checked by reading, not compiled, not run on hardware
//
#include <hip/hip_runtime.h>
#include <hip/hip_bf16.h>

#define N_NODES 8192
#define F_IN    512
#define F_OUT   64
#define NSEG    4
#define SEGLEN  (N_NODES / NSEG)     // 2048 columns per segment
#define GAT_ALPHA 0.2f
#define LOG2E 1.44269504088896340736f

typedef __attribute__((ext_vector_type(16))) __bf16 v16bf;
typedef __attribute__((ext_vector_type(8)))  float  v8f;
typedef __attribute__((ext_vector_type(2)))  float  v2f;

union BFrag { v16bf v; uint4 q[2]; };

__device__ __forceinline__ unsigned bf16_rne(float f) {
    unsigned u = __float_as_uint(f);
    return (u + 0x7FFFu + ((u >> 16) & 1u)) >> 16;
}
__device__ __forceinline__ unsigned pk_bf16(float lo, float hi) {
    return bf16_rne(lo) | (bf16_rne(hi) << 16);
}

// ---------------------------------------------------------------------------
// Pack row-major f32 [nC32*32 x 64] into bf16 WMMA-B lane layout.
// ---------------------------------------------------------------------------
__global__ void pack_b_kernel(const float* __restrict__ src,
                              unsigned* __restrict__ dst, int nC32) {
    int g = blockIdx.x * blockDim.x + threadIdx.x;
    if (g >= nC32 * 1024) return;
    int v = g & 7, lane = (g >> 3) & 31, t = (g >> 8) & 3, c32 = g >> 10;
    int K   = c32 * 32 + (lane >> 4) * 8 + 2 * (v & 3) + (v >> 2) * 16;
    int col = t * 16 + (lane & 15);
    dst[g] = pk_bf16(src[(size_t)K * F_OUT + col], src[(size_t)(K + 1) * F_OUT + col]);
}

// ---------------------------------------------------------------------------
// Pack W (512x64 f32) into f32 WMMA-B lane layout for 16x16x4_f32.
// ---------------------------------------------------------------------------
__global__ void pack_b_f32_kernel(const float* __restrict__ src,
                                  float* __restrict__ dst) {
    int g = blockIdx.x * blockDim.x + threadIdx.x;   // 32768 total
    int v = g & 1, lane = (g >> 1) & 31, t = (g >> 6) & 3, c4 = g >> 8;
    int K   = c4 * 4 + (lane >> 4) * 2 + v;
    int col = t * 16 + (lane & 15);
    dst[g] = src[(size_t)K * F_OUT + col];
}

// ---------------------------------------------------------------------------
// Kernel 1: Wh = h @ W in full fp32 via v_wmma_f32_16x16x4_f32.
// ---------------------------------------------------------------------------
__global__ void __launch_bounds__(256)
gemm_wh_kernel(const float* __restrict__ h, const float* __restrict__ WBf,
               float* __restrict__ Wh) {
    int lane = threadIdx.x & 31;
    int wid  = threadIdx.x >> 5;
    int tile = blockIdx.x * 8 + wid;      // 2048 tiles
    int m0   = (tile >> 2) * 16;
    int t    = tile & 3;
    int mrow = lane & 15;
    int kq   = lane >> 4;
    const float* hrow = h + (size_t)(m0 + mrow) * F_IN;
    v8f acc = {};
    #pragma unroll 8
    for (int k = 0; k < F_IN; k += 4) {
        float2 hv = *(const float2*)(hrow + k + kq * 2);
        v2f a; a[0] = hv.x; a[1] = hv.y;
        float2 bv = *(const float2*)(WBf + (((size_t)(k >> 2) * 4 + t) * 32 + lane) * 2);
        v2f b; b[0] = bv.x; b[1] = bv.y;
        acc = __builtin_amdgcn_wmma_f32_16x16x4_f32(false, a, false, b,
                                                    (short)0, acc, false, false);
    }
    #pragma unroll
    for (int k = 0; k < 8; ++k) {
        int row = m0 + k + ((lane < 16) ? 0 : 8);
        int col = t * 16 + (lane & 15);
        Wh[(size_t)row * F_OUT + col] = acc[k];
    }
}

// ---------------------------------------------------------------------------
// Kernel 2a: s1 = (Wh@a1)*log2(e), s2 = (Wh@a2)*log2(e)  (base-2 softmax domain)
// ---------------------------------------------------------------------------
__global__ void svec_kernel(const float* __restrict__ Wh, const float* __restrict__ a,
                            float* __restrict__ s1, float* __restrict__ s2) {
    int row = blockIdx.x * blockDim.x + threadIdx.x;
    const float4* wr = (const float4*)(Wh + (size_t)row * F_OUT);
    const float4* a1 = (const float4*)(a);
    const float4* a2 = (const float4*)(a + F_OUT);
    float x1 = 0.f, x2 = 0.f;
    #pragma unroll
    for (int i = 0; i < 16; ++i) {
        float4 w = wr[i], c1 = a1[i], c2 = a2[i];
        x1 += w.x * c1.x + w.y * c1.y + w.z * c1.z + w.w * c1.w;
        x2 += w.x * c2.x + w.y * c2.y + w.z * c2.z + w.w * c2.w;
    }
    s1[row] = x1 * LOG2E;
    s2[row] = x2 * LOG2E;
}

// ---------------------------------------------------------------------------
// Kernel 2b: global max of s2 (single block)
// ---------------------------------------------------------------------------
__global__ void s2max_kernel(const float* __restrict__ s2, float* __restrict__ s2max) {
    __shared__ float red[256];
    float m = -3.4e38f;
    for (int i = threadIdx.x; i < N_NODES; i += 256) m = fmaxf(m, s2[i]);
    red[threadIdx.x] = m;
    __syncthreads();
    for (int s = 128; s > 0; s >>= 1) {
        if (threadIdx.x < s) red[threadIdx.x] = fmaxf(red[threadIdx.x], red[threadIdx.x + s]);
        __syncthreads();
    }
    if (threadIdx.x == 0) s2max[0] = red[0];
}

// ---------------------------------------------------------------------------
// Kernel 3: split-K fused mask + softmax-numerator + P@Wh with a STATIC
// per-row shift M_r = lrelu(s1_r + max(s2)) (tight upper bound on the row max
// since lrelu is monotone).  No online rescaling, no shuffle reductions; the
// softmax denominator comes from a ones-B WMMA (accL).  Partials just add.
// ---------------------------------------------------------------------------
__global__ void __launch_bounds__(256)
gat_attn_kernel(const int* __restrict__ adj, const float* __restrict__ s1,
                const float* __restrict__ s2, const float* __restrict__ s2max,
                const unsigned* __restrict__ WhB,
                float* __restrict__ accP, float* __restrict__ lP) {
    __shared__ uint2 Pt[8][16 * 32];      // per-wave 16x128 bf16 tile
    int lane = threadIdx.x & 31;
    int wid  = threadIdx.x >> 5;
    int tid  = blockIdx.x * 8 + wid;      // 2048 wave-tiles
    int row0 = (tid & (N_NODES / 16 - 1)) * 16;
    int seg  = tid >> 9;                  // 0..3
    int segStart = seg * SEGLEN;
    bool lo16 = lane < 16;

    float s2m = s2max[0];
    float s1r[16], Mr[16];
    #pragma unroll
    for (int r = 0; r < 16; ++r) {
        s1r[r] = s1[row0 + r];
        float tmax = s1r[r] + s2m;
        Mr[r] = fmaxf(tmax, GAT_ALPHA * tmax);   // lrelu of the bound
    }
    v8f acc[4] = {{}, {}, {}, {}};
    v8f accL = {};                        // row sums via ones-WMMA
    BFrag Bones;
    Bones.q[0] = make_uint4(0x3F803F80u, 0x3F803F80u, 0x3F803F80u, 0x3F803F80u);
    Bones.q[1] = Bones.q[0];
    uint2* myP = &Pt[wid][0];

    for (int j0 = segStart; j0 < segStart + SEGLEN; j0 += 128) {
        int jn = (j0 + 128 < segStart + SEGLEN) ? j0 + 128 : segStart;
        float4 sv = ((const float4*)(s2 + j0))[lane];
        #pragma unroll
        for (int r = 0; r < 16; ++r) {
            const int* arow = adj + (size_t)(row0 + r) * N_NODES;
            int4 av = ((const int4*)(arow + j0))[lane];
            __builtin_prefetch((const char*)(arow + jn) + lane * 16, 0, 1);
            float b = s1r[r], M = Mr[r];
            float e0 = b + sv.x; e0 = fmaxf(e0, GAT_ALPHA * e0);
            float e1 = b + sv.y; e1 = fmaxf(e1, GAT_ALPHA * e1);
            float e2 = b + sv.z; e2 = fmaxf(e2, GAT_ALPHA * e2);
            float e3 = b + sv.w; e3 = fmaxf(e3, GAT_ALPHA * e3);
            float p0 = av.x > 0 ? __builtin_amdgcn_exp2f(e0 - M) : 0.f;
            float p1 = av.y > 0 ? __builtin_amdgcn_exp2f(e1 - M) : 0.f;
            float p2 = av.z > 0 ? __builtin_amdgcn_exp2f(e2 - M) : 0.f;
            float p3 = av.w > 0 ? __builtin_amdgcn_exp2f(e3 - M) : 0.f;
            myP[r * 32 + lane] = make_uint2(pk_bf16(p0, p1), pk_bf16(p2, p3));
        }
        asm volatile("s_wait_dscnt 0" ::: "memory");   // same-wave DS in-order; fence vs reorder

        int c32base = j0 >> 5;
        int m   = lane & 15;
        int kh2 = (lane >> 4) * 16;
        const char* base = (const char*)myP;
        #pragma unroll
        for (int ks = 0; ks < 4; ++ks) {
            BFrag A;   // WMMA-A layout straight out of the row-major LDS tile
            A.q[0] = *(const uint4*)(base + m * 256 + ks * 64 + kh2);
            A.q[1] = *(const uint4*)(base + m * 256 + ks * 64 + kh2 + 32);
            #pragma unroll
            for (int t = 0; t < 4; ++t) {
                const uint4* bp = (const uint4*)(WhB +
                    (((size_t)(c32base + ks) * 4 + t) * 32 + lane) * 8);
                BFrag B;
                B.q[0] = bp[0];
                B.q[1] = bp[1];
                acc[t] = __builtin_amdgcn_wmma_f32_16x16x32_bf16(
                    false, A.v, false, B.v, (short)0, acc[t], false, false);
            }
            accL = __builtin_amdgcn_wmma_f32_16x16x32_bf16(
                false, A.v, false, Bones.v, (short)0, accL, false, false);
        }
    }

    // write partials: acc (unnormalized) and row sums; same shift => plain add later
    size_t segAcc = (size_t)seg * N_NODES * F_OUT;
    #pragma unroll
    for (int t = 0; t < 4; ++t)
        #pragma unroll
        for (int k = 0; k < 8; ++k) {
            int row = row0 + k + (lo16 ? 0 : 8);
            int col = t * 16 + (lane & 15);
            accP[segAcc + (size_t)row * F_OUT + col] = acc[t][k];
        }
    if ((lane & 15) == 0) {
        #pragma unroll
        for (int k = 0; k < 8; ++k)
            lP[seg * N_NODES + row0 + k + (lo16 ? 0 : 8)] = accL[k];
    }
}

// ---------------------------------------------------------------------------
// Kernel 4: sum split-K partials, normalize, ELU.
// ---------------------------------------------------------------------------
__global__ void combine_kernel(const float* __restrict__ accP,
                               const float* __restrict__ lP,
                               float* __restrict__ out) {
    int g   = blockIdx.x * blockDim.x + threadIdx.x;    // 8192*64
    int row = g >> 6;
    float L = lP[row] + lP[N_NODES + row] + lP[2 * N_NODES + row] + lP[3 * N_NODES + row];
    const size_t S = (size_t)N_NODES * F_OUT;
    float A = accP[g] + accP[S + g] + accP[2 * S + g] + accP[3 * S + g];
    float v = A / fmaxf(L, 1e-30f);
    v = v > 0.f ? v : (__expf(v) - 1.f);                // ELU
    out[g] = v;
}

// ---------------------------------------------------------------------------
extern "C" void kernel_launch(void* const* d_in, const int* in_sizes, int n_in,
                              void* d_out, int out_size, void* d_ws, size_t ws_size,
                              hipStream_t stream) {
    const float* h   = (const float*)d_in[0];
    const int*   adj = (const int*)d_in[1];
    const float* W   = (const float*)d_in[2];
    const float* a   = (const float*)d_in[3];
    float* out = (float*)d_out;

    char* ws = (char*)d_ws;
    const size_t MB = 1024 * 1024;
    float*    Wh    = (float*)ws;                           // 2 MB
    unsigned* WhB   = (unsigned*)(ws + 2 * MB);             // 1 MB
    float*    s1    = (float*)(ws + 3 * MB);                // 32 KB
    float*    s2    = s1 + N_NODES;                         // 32 KB
    float*    WBf   = (float*)(ws + 3 * MB + 64 * 1024);    // 128 KB
    float*    s2mx  = (float*)(ws + 3 * MB + 192 * 1024);   // 4 B
    float*    accP  = (float*)(ws + 4 * MB);                // 8 MB (segment partials)
    float*    lP    = (float*)(ws + 12 * MB);               // 128 KB

    // 0) pack W into f32 WMMA-B layout
    pack_b_f32_kernel<<<128, 256, 0, stream>>>(W, WBf);
    // 1) Wh = h @ W, fp32 WMMA
    gemm_wh_kernel<<<256, 256, 0, stream>>>(h, WBf, Wh);
    // 2a) attention vectors (pre-scaled by log2 e)
    svec_kernel<<<N_NODES / 256, 256, 0, stream>>>(Wh, a, s1, s2);
    // 2b) global max of s2 for the static softmax shift
    s2max_kernel<<<1, 256, 0, stream>>>(s2, s2mx);
    // 2c) pack Wh into bf16 WMMA-B layout
    pack_b_kernel<<<1024, 256, 0, stream>>>(Wh, WhB, 256);
    // 3) fused masked softmax-numerator attention, 4 column segments
    gat_attn_kernel<<<256, 256, 0, stream>>>(adj, s1, s2, s2mx, WhB, accP, lP);
    // 4) sum partials + normalize + ELU
    combine_kernel<<<(N_NODES * F_OUT) / 256, 256, 0, stream>>>(accP, lP, out);
}
